// Conv2dQuant_5368709120690
// MI455X (gfx1250) — matrix operand order repaired and checked by reading, not compiled, not run on hardware
//
#include <hip/hip_runtime.h>

typedef __attribute__((ext_vector_type(16))) _Float16 v16h;
typedef __attribute__((ext_vector_type(8)))  float    v8f;

#define C_IN       64
#define KDIM       576            // C_IN * 3 * 3
#define L_OUT      784            // 28 * 28
#define O_OUT      64
#define NTILES     49             // 784 / 16
#define LDS_STRIDE 584            // 576 + 8 halves: conflict-free column stride

__global__ __launch_bounds__(128)
void conv2d_quant_wmma(const float* __restrict__ X,
                       const float* __restrict__ W,
                       const float* __restrict__ Bias,
                       float* __restrict__ Out)
{
    __shared__ _Float16 lds_patch[16 * LDS_STRIDE];  // 16 cols x K, column-major

    const int b     = blockIdx.x / NTILES;
    const int ntile = blockIdx.x % NTILES;
    const int tid   = threadIdx.x;
    const int lane  = tid & 31;
    const int wid   = tid >> 5;           // 4 waves -> 4 M tiles of 16 rows

    // Warm L2 with the (heavily reused) weight panel: global_prefetch_b8.
    __builtin_prefetch(W + tid * 64, 0, 1);

    // ---- Stage im2col patch tile (K=576 x N=16) into LDS, column-major ----
    // 16*576 / 128 threads = 72 elements per thread, fully uniform (no EXEC
    // divergence before the WMMAs).
    for (int idx = tid; idx < 16 * KDIM; idx += 128) {
        const int n  = idx / KDIM;
        const int k  = idx - n * KDIM;
        const int l  = ntile * 16 + n;
        const int oh = l / 28, ow = l - (l / 28) * 28;
        const int c  = k / 9,  r  = k - c * 9;
        const int ih = oh - 1 + r / 3;
        const int iw = ow - 1 + (r % 3);
        float v = 0.0f;
        if ((unsigned)ih < 28u && (unsigned)iw < 28u)
            v = X[((b * C_IN + c) * 28 + ih) * 28 + iw];
        lds_patch[n * LDS_STRIDE + k] = (_Float16)v;
    }
    __syncthreads();

    // ---- Per-wave 16x16 output tile, K loop of 18 x (16x16x32) WMMAs ----
    const int mtile = wid;
    const int mrow  = lane & 15;
    const int khalf = lane >> 4;          // 0 or 1: selects the K half-chunks
    // A (16x32 f16) lane layout: lane<16 holds K {0..15} of row M=lane,
    // lane>=16 holds K {16..31} of row M=lane-16.
    const float*    wrow = W + (mtile * 16 + mrow) * KDIM + khalf * 16;
    // B (32x16 f16) lane layout: 16 contiguous K of column n, kbase = khalf*16.
    const _Float16* bcol = lds_patch + (lane & 15) * LDS_STRIDE + khalf * 16;

    v8f acc = {};
#pragma unroll 2
    for (int s = 0; s < 18; ++s) {
        const float* wp = wrow + s * 32;          // 32B-aligned
        v16h a;
#pragma unroll
        for (int e = 0; e < 8; ++e) a[e]     = (_Float16)wp[e];
#pragma unroll
        for (int e = 0; e < 8; ++e) a[8 + e] = (_Float16)wp[8 + e];

        const _Float16* bp = bcol + s * 32;       // 16B-aligned in LDS
        v16h bm;
#pragma unroll
        for (int e = 0; e < 16; ++e) bm[e] = bp[e];

        // D = A x B + C  (f32 accumulate)
        acc = __builtin_amdgcn_wmma_f32_16x16x32_f16(
                  /*neg_a=*/false, a, /*neg_b=*/false, bm,
                  /*c_mod=*/(short)0, acc, /*reuse_a=*/false, /*reuse_b=*/false);
    }

    // ---- Exact sum-level + bias-level quantization epilogue ----
    // C/D layout: n = lane&15 ; VGPR r holds M = (lane>>4)*8 + r.
    const int n = lane & 15;
    const int l = ntile * 16 + n;
#pragma unroll
    for (int r = 0; r < 8; ++r) {
        const int m = khalf * 8 + r;
        const int o = mtile * 16 + m;
        float v  = acc[r];
        // q(v) = clip(rne(8v), -128, 127) / 8   (rintf = round half-to-even)
        float q1 = fminf(fmaxf(rintf(v * 8.0f), -128.0f), 127.0f) * 0.125f;
        float q2 = fminf(fmaxf(rintf((q1 + Bias[o]) * 8.0f), -128.0f), 127.0f) * 0.125f;
        Out[(b * O_OUT + o) * L_OUT + l] = q2;
    }
}

extern "C" void kernel_launch(void* const* d_in, const int* in_sizes, int n_in,
                              void* d_out, int out_size, void* d_ws, size_t ws_size,
                              hipStream_t stream) {
    const float* X  = (const float*)d_in[0];   // [4,64,28,28]
    const float* W  = (const float*)d_in[1];   // [64,64,3,3]
    const float* Bi = (const float*)d_in[2];   // [64]
    float*       O  = (float*)d_out;           // [4,64,28,28]

    dim3 grid(4 * NTILES);   // one block per (batch, 16-wide N tile)
    dim3 block(128);         // 4 waves, one 16x16 M tile each
    hipLaunchKernelGGL(conv2d_quant_wmma, grid, block, 0, stream, X, W, Bi, O);
}
